// GptOssExperts_45586782880009
// MI455X (gfx1250) — matrix operand order repaired and checked by reading, not compiled, not run on hardware
//
#include <hip/hip_runtime.h>

// ---------------------------------------------------------------------------
// GPT-OSS MoE experts for MI455X (gfx1250) — bf16 WMMA grouped GEMM,
// double-buffered LDS pipeline with async-DMA activation tiles.
//   TOKENS=512, TOP_K=4 -> 2048 pairs; HIDDEN=INTER=2880; 8 experts.
//   GEMM1: h = W_gu[e] (5760x2880) @ t -> clamp/SwiGLU -> u (2880, bf16)
//   GEMM2: y = W_d [e] (2880x2880) @ u + bias            (f32 out)
// Roofline: 800MB fp32 weights @ 23.3TB/s ~= 34us floor; bf16 WMMA compute
// ~20us for 102 GFLOP -> HBM-bound. Weights convert f32->bf16 in registers;
// the pipeline keeps the weight stream and the TDM/async path busy under
// the WMMA block every iteration.
// ---------------------------------------------------------------------------

typedef __bf16 bf16_t;
typedef bf16_t v16bf __attribute__((ext_vector_type(16)));
typedef float  v8f   __attribute__((ext_vector_type(8)));

union Frag32B { uint4 u4[2]; v16bf v; };

#define NUM_EXPERTS 8
#define TOKENS      512
#define TOP_K       4
#define NPAIRS      (TOKENS * TOP_K)      // 2048
#define HIDDEN      2880
#define INTER       2880
#define GU_ROWS     (2 * INTER)           // 5760
#define LIMIT       7.0f

#define TILE_M 256
#define TILE_N 64
#define TILE_K 64
#define KSTRIDE 72                        // padded LDS row stride (bf16), 144B
#define NKSTEPS (HIDDEN / TILE_K)         // 45
#define A_BUF_ELEMS (TILE_M * KSTRIDE)    // 18432 (36 KB)
#define B_BUF_ELEMS (TILE_N * KSTRIDE)    //  4608 (9 KB)

// ---- CDNA5 async DMA: global -> LDS, tracked by ASYNCcnt ------------------
__device__ __forceinline__ void async_copy_b128(unsigned lds_off, const void* g) {
    unsigned long long ga = (unsigned long long)g;
    asm volatile("global_load_async_to_lds_b128 %0, %1, off"
                 :: "v"(lds_off), "v"(ga) : "memory");
}
__device__ __forceinline__ void wait_async0() {
    asm volatile("s_wait_asynccnt 0x0" ::: "memory");
}

// ---------------------------------------------------------------------------
// Kernel 1: routing — per-expert contiguous pair lists.
// ---------------------------------------------------------------------------
__global__ __launch_bounds__(256) void route_kernel(
    const int* __restrict__ ei, int* __restrict__ counts,
    int* __restrict__ offsets, int* __restrict__ list)
{
    __shared__ int cnt[NUM_EXPERTS];
    __shared__ int offs[NUM_EXPERTS];
    const int tid = threadIdx.x;
    if (tid < NUM_EXPERTS) cnt[tid] = 0;
    __syncthreads();
    for (int p = tid; p < NPAIRS; p += 256) atomicAdd(&cnt[ei[p]], 1);
    __syncthreads();
    if (tid == 0) {
        int acc = 0;
        for (int e = 0; e < NUM_EXPERTS; ++e) {
            offs[e] = acc; offsets[e] = acc; counts[e] = cnt[e]; acc += cnt[e];
        }
    }
    __syncthreads();
    if (tid < NUM_EXPERTS) cnt[tid] = 0;
    __syncthreads();
    for (int p = tid; p < NPAIRS; p += 256) {
        const int e = ei[p];
        const int pos = atomicAdd(&cnt[e], 1);
        list[offs[e] + pos] = p;
    }
}

// ---------------------------------------------------------------------------
// Kernel 2: t (f32) -> bf16 once per launch.
// ---------------------------------------------------------------------------
__global__ __launch_bounds__(256) void convert_t_kernel(
    const float* __restrict__ t, bf16_t* __restrict__ tb, int n)
{
    const int i = (blockIdx.x * 256 + threadIdx.x) * 4;
    if (i + 3 < n) {
        const float4 f = *reinterpret_cast<const float4*>(t + i);
        tb[i + 0] = (bf16_t)f.x; tb[i + 1] = (bf16_t)f.y;
        tb[i + 2] = (bf16_t)f.z; tb[i + 3] = (bf16_t)f.w;
    }
}

// ---------------------------------------------------------------------------
// Shared GEMM core: double-buffered, async A / reg-staged B, 16 WMMA / step.
// A rows gathered through 'arow' pointers (precomputed per thread).
// Produces acc[2][4] (wave covers 32 rows x 64 cols).
// ---------------------------------------------------------------------------
struct GemmCtx {
    const bf16_t* asrc[8];   // per-thread A source (excl. k offset)
    unsigned      alds[8];   // per-thread A LDS byte offset within buffer
    const float*  bsrc[4];   // per-thread B source (excl. k offset)
    bf16_t*       bdst[4];   // per-thread B LDS dst within buffer
};

template <typename EpilogueT>
__device__ __forceinline__ void gemm_core(
    bf16_t* As, bf16_t* Bs, const GemmCtx& cx,
    int wave, int lhalf, int l15, v8f (&acc)[2][4])
{
    float4 breg[4];
    // ---- prologue: tile 0 in flight ---------------------------------------
    #pragma unroll
    for (int s = 0; s < 8; ++s)
        async_copy_b128((unsigned)(size_t)As + cx.alds[s], cx.asrc[s]);
    #pragma unroll
    for (int s = 0; s < 4; ++s)
        breg[s] = *reinterpret_cast<const float4*>(cx.bsrc[s]);

    // fragment LDS element offsets (invariant; buffer toggled by base ptr)
    int aoff[2], boff[4];
    #pragma unroll
    for (int i = 0; i < 2; ++i)
        aoff[i] = (wave * 32 + i * 16 + l15) * KSTRIDE + lhalf * 8;
    #pragma unroll
    for (int j = 0; j < 4; ++j)
        boff[j] = (j * 16 + l15) * KSTRIDE + lhalf * 8;

    int cur = 0;
    for (int it = 0; it < NKSTEPS; ++it) {
        bf16_t* Asc = As + cur * A_BUF_ELEMS;
        bf16_t* Bsc = Bs + cur * B_BUF_ELEMS;
        const int kn = (it + 1) * TILE_K;

        // 1. commit B(it) regs -> LDS (cvt f32->bf16)
        #pragma unroll
        for (int s = 0; s < 4; ++s) {
            bf16_t* dst = cx.bdst[s] + cur * B_BUF_ELEMS;
            dst[0] = (bf16_t)breg[s].x; dst[1] = (bf16_t)breg[s].y;
            dst[2] = (bf16_t)breg[s].z; dst[3] = (bf16_t)breg[s].w;
        }
        // 2. prefetch B(it+1) into regs (overlaps compute below)
        if (it + 1 < NKSTEPS) {
            #pragma unroll
            for (int s = 0; s < 4; ++s)
                breg[s] = *reinterpret_cast<const float4*>(cx.bsrc[s] + kn);
        }
        // 3. drain A(it) DMA, make LDS visible
        wait_async0();
        __syncthreads();
        // 4. issue A(it+1) DMA into the other buffer (all readers are past
        //    the barrier, so overwriting As[nxt] is safe; completes under
        //    the WMMA block below and is drained next iteration).
        if (it + 1 < NKSTEPS) {
            const unsigned nbase =
                (unsigned)(size_t)As + (unsigned)((cur ^ 1) * A_BUF_ELEMS * (int)sizeof(bf16_t));
            #pragma unroll
            for (int s = 0; s < 8; ++s)
                async_copy_b128(nbase + cx.alds[s], cx.asrc[s] + kn);
        }
        // 5. 16 WMMAs on the current buffers
        #pragma unroll
        for (int kk = 0; kk < TILE_K; kk += 32) {
            Frag32B a[2], b[4];
            #pragma unroll
            for (int i = 0; i < 2; ++i) {
                a[i].u4[0] = *reinterpret_cast<const uint4*>(Asc + aoff[i] + kk);
                a[i].u4[1] = *reinterpret_cast<const uint4*>(Asc + aoff[i] + kk + 16);
            }
            #pragma unroll
            for (int j = 0; j < 4; ++j) {
                b[j].u4[0] = *reinterpret_cast<const uint4*>(Bsc + boff[j] + kk);
                b[j].u4[1] = *reinterpret_cast<const uint4*>(Bsc + boff[j] + kk + 16);
            }
            #pragma unroll
            for (int i = 0; i < 2; ++i)
                #pragma unroll
                for (int j = 0; j < 4; ++j)
                    acc[i][j] = __builtin_amdgcn_wmma_f32_16x16x32_bf16(
                        false, a[i].v, false, b[j].v, (short)0, acc[i][j], false, false);
        }
        cur ^= 1;
    }
}

// ---------------------------------------------------------------------------
// Kernel 3: grouped GEMM1 (gate_up) + fused clamp/SwiGLU via lane pairing.
// grid = (8 mtiles, 90 ntiles, 8 experts), block = 256 threads (8 waves).
// ---------------------------------------------------------------------------
__global__ __launch_bounds__(256) void gemm1_gateup_kernel(
    const bf16_t* __restrict__ tb,     // [TOKENS][HIDDEN] bf16
    const float*  __restrict__ w,      // [8][GU_ROWS][HIDDEN] f32
    const float*  __restrict__ bias,   // [8][GU_ROWS] f32
    const int* __restrict__ counts, const int* __restrict__ offsets,
    const int* __restrict__ list,
    bf16_t* __restrict__ u)            // [NPAIRS][INTER] bf16
{
    const int e = blockIdx.z;
    const int cnt = counts[e];
    const int mbase = blockIdx.x * TILE_M;
    if (mbase >= cnt) return;
    const int n0 = blockIdx.y * TILE_N;
    const int off = offsets[e];

    __shared__ bf16_t As[2 * A_BUF_ELEMS];   // 72 KB
    __shared__ bf16_t Bs[2 * B_BUF_ELEMS];   // 18 KB
    __shared__ int rowPairS[TILE_M];

    const int tid = threadIdx.x;
    {
        const int idx = mbase + tid;
        rowPairS[tid] = (idx < cnt) ? list[off + idx] : list[off];
    }
    __syncthreads();

    const int lane  = tid & 31;
    const int wave  = tid >> 5;
    const int lhalf = lane >> 4;
    const int l15   = lane & 15;
    const float* wbase = w + (size_t)e * GU_ROWS * HIDDEN;

    // ---- precompute all per-thread copy addressing (loop-invariant) -------
    GemmCtx cx;
    #pragma unroll
    for (int s = 0; s < 8; ++s) {
        const int unit = tid + 256 * s;              // 2048 x 16B units
        const int r = unit >> 3;
        const int c = (unit & 7) * 8;
        const int tok = rowPairS[r] >> 2;            // pair -> token
        cx.asrc[s] = tb + (size_t)tok * HIDDEN + c;
        cx.alds[s] = (unsigned)((r * KSTRIDE + c) * (int)sizeof(bf16_t));
    }
    #pragma unroll
    for (int s = 0; s < 4; ++s) {
        const int unit = tid + 256 * s;              // 1024 x float4 units
        const int r = unit >> 4;
        const int c = (unit & 15) * 4;
        cx.bsrc[s] = wbase + (size_t)(n0 + r) * HIDDEN + c;
        cx.bdst[s] = Bs + r * KSTRIDE + c;
    }

    v8f acc[2][4] = {};
    gemm_core<int>(As, Bs, cx, wave, lhalf, l15, acc);

    // ---- epilogue: gate/lin live in adjacent lanes -> shfl_xor pairing ----
    const float* bptr = bias + (size_t)e * GU_ROWS + n0;
    float bg[4], bl[4];
    #pragma unroll
    for (int j = 0; j < 4; ++j) {
        const int N  = j * 16 + l15;
        const int Np = (l15 & 1) ? N : N + 1;        // odd lanes: dummy
        bg[j] = bptr[N];
        bl[j] = bptr[Np];
    }
    #pragma unroll
    for (int i = 0; i < 2; ++i)
        #pragma unroll
        for (int j = 0; j < 4; ++j)
            #pragma unroll
            for (int r = 0; r < 8; ++r) {
                const float val = acc[i][j][r];
                const float partner = __shfl_xor(val, 1, 32);
                const int m = wave * 32 + i * 16 + lhalf * 8 + r;
                if (((l15 & 1) == 0) && (mbase + m < cnt)) {
                    float g = fminf(val + bg[j], LIMIT);
                    float l = fminf(fmaxf(partner + bl[j], -LIMIT), LIMIT);
                    const float glu = g / (1.0f + __expf(-1.702f * g));
                    const float uval = glu * (l + 1.0f);
                    const int p = rowPairS[m];
                    const int uc = (n0 + j * 16 + l15) >> 1;
                    u[(size_t)p * INTER + uc] = (bf16_t)uval;
                }
            }
}

// ---------------------------------------------------------------------------
// Kernel 4: grouped GEMM2 (down-proj) -> f32 output + bias.
// grid = (8 mtiles, 45 ntiles, 8 experts), block = 256.
// ---------------------------------------------------------------------------
__global__ __launch_bounds__(256) void gemm2_down_kernel(
    const bf16_t* __restrict__ ub,     // [NPAIRS][INTER] bf16
    const float*  __restrict__ w,      // [8][HIDDEN][INTER] f32
    const float*  __restrict__ bias,   // [8][HIDDEN] f32
    const int* __restrict__ counts, const int* __restrict__ offsets,
    const int* __restrict__ list,
    float* __restrict__ out)           // [NPAIRS][HIDDEN] f32
{
    const int e = blockIdx.z;
    const int cnt = counts[e];
    const int mbase = blockIdx.x * TILE_M;
    if (mbase >= cnt) return;
    const int n0 = blockIdx.y * TILE_N;
    const int off = offsets[e];

    __shared__ bf16_t As[2 * A_BUF_ELEMS];
    __shared__ bf16_t Bs[2 * B_BUF_ELEMS];
    __shared__ int rowPairS[TILE_M];

    const int tid = threadIdx.x;
    {
        const int idx = mbase + tid;
        rowPairS[tid] = (idx < cnt) ? list[off + idx] : list[off];
    }
    __syncthreads();

    const int lane  = tid & 31;
    const int wave  = tid >> 5;
    const int lhalf = lane >> 4;
    const int l15   = lane & 15;
    const float* wbase = w + (size_t)e * HIDDEN * INTER;

    GemmCtx cx;
    #pragma unroll
    for (int s = 0; s < 8; ++s) {
        const int unit = tid + 256 * s;
        const int r = unit >> 3;
        const int c = (unit & 7) * 8;
        const int p = rowPairS[r];                   // pair indexes u directly
        cx.asrc[s] = ub + (size_t)p * INTER + c;
        cx.alds[s] = (unsigned)((r * KSTRIDE + c) * (int)sizeof(bf16_t));
    }
    #pragma unroll
    for (int s = 0; s < 4; ++s) {
        const int unit = tid + 256 * s;
        const int r = unit >> 4;
        const int c = (unit & 15) * 4;
        cx.bsrc[s] = wbase + (size_t)(n0 + r) * INTER + c;
        cx.bdst[s] = Bs + r * KSTRIDE + c;
    }

    v8f acc[2][4] = {};
    gemm_core<int>(As, Bs, cx, wave, lhalf, l15, acc);

    float bv[4];
    #pragma unroll
    for (int j = 0; j < 4; ++j)
        bv[j] = bias[(size_t)e * HIDDEN + n0 + j * 16 + l15];

    #pragma unroll
    for (int i = 0; i < 2; ++i)
        #pragma unroll
        for (int j = 0; j < 4; ++j)
            #pragma unroll
            for (int r = 0; r < 8; ++r) {
                const int m = wave * 32 + i * 16 + lhalf * 8 + r;
                if (mbase + m < cnt) {
                    const int N = n0 + j * 16 + l15;
                    const int p = rowPairS[m];
                    out[(size_t)p * HIDDEN + N] = acc[i][j][r] + bv[j];
                }
            }
}

// ---------------------------------------------------------------------------
// Host-side launcher.
// ---------------------------------------------------------------------------
extern "C" void kernel_launch(void* const* d_in, const int* in_sizes, int n_in,
                              void* d_out, int out_size, void* d_ws, size_t ws_size,
                              hipStream_t stream) {
    const float* t   = (const float*)d_in[0];
    const int*   ei  = (const int*)  d_in[1];
    const float* gup = (const float*)d_in[2];
    const float* gub = (const float*)d_in[3];
    const float* dwn = (const float*)d_in[4];
    const float* dbn = (const float*)d_in[5];
    float* out = (float*)d_out;

    char* ws = (char*)d_ws;
    int*    counts  = (int*)(ws + 0);
    int*    offsets = (int*)(ws + 64);
    int*    list    = (int*)(ws + 128);                       // 2048 ints
    bf16_t* tb      = (bf16_t*)(ws + 16384);                  // 2.95 MB
    bf16_t* u       = (bf16_t*)(ws + 16384 + (size_t)TOKENS * HIDDEN * 2); // 11.8 MB

    route_kernel<<<1, 256, 0, stream>>>(ei, counts, offsets, list);

    const int tElems = TOKENS * HIDDEN;
    convert_t_kernel<<<tElems / 4 / 256, 256, 0, stream>>>(t, tb, tElems);

    dim3 g1(NPAIRS / TILE_M, GU_ROWS / TILE_N, NUM_EXPERTS);  // (8, 90, 8)
    gemm1_gateup_kernel<<<g1, 256, 0, stream>>>(tb, gup, gub, counts, offsets, list, u);

    dim3 g2(NPAIRS / TILE_M, HIDDEN / TILE_N, NUM_EXPERTS);   // (8, 45, 8)
    gemm2_down_kernel<<<g2, 256, 0, stream>>>(u, dwn, dbn, counts, offsets, list, out);
}
